// GRN_87625922773466
// MI455X (gfx1250) — compile-verified, matrix operand order checked by reading, not run on hardware
//
#include <hip/hip_runtime.h>
#include <hip/hip_bf16.h>

// ---------------------------------------------------------------------------
// AttentiveFP-style GNN on MI455X (gfx1250), wave32 + v_wmma_f32_16x16x32_f16.
// ~200 GFLOP of GEMMs vs ~0.25GB traffic -> compute bound -> matrix pipe,
// f16 inputs / f32 accumulate. Weights padded+converted to f16 once per launch.
// GEMM: templated KSTEPS/MODE -> branch-free full unroll; 4 M-tiles per wave
// so each weight (B) fragment feeds 4 WMMAs and loads pipeline across steps.
// ---------------------------------------------------------------------------

typedef __attribute__((ext_vector_type(16))) _Float16 v16h;
typedef __attribute__((ext_vector_type(8)))  float    v8f;

#define PF   224   // padded F (200 -> 7 K-steps of 32)
#define F_   200
#define LDIM 256
#define BDIM 128
#define NNB  8
#define M_ATOMS (BDIM * LDIM)        // 32768
#define M_BONDS (M_ATOMS * NNB)      // 262144

union FragU { float4 f4[2]; v16h v; };

// 16-bit A/B fragment: lane holds halves [k0..k0+7] and [k0+16..k0+23]
__device__ __forceinline__ v16h load_frag(const _Float16* p) {
  FragU u;
  u.f4[0] = *(const float4*)(p);
  u.f4[1] = *(const float4*)(p + 16);
  return u.v;
}

__device__ __forceinline__ v8f wmma_f16(v16h a, v16h b, v8f c) {
  return __builtin_amdgcn_wmma_f32_16x16x32_f16(false, a, false, b, (short)0, c,
                                                false, false);
}

__device__ __forceinline__ float sigm(float x) { return 1.f / (1.f + expf(-x)); }
__device__ __forceinline__ float eluf(float x) { return x > 0.f ? x : (expf(x) - 1.f); }

// --------------------------- weight conversion -----------------------------
// dst [Npad, Kpad] f16, segmented padding: padded col p -> seg=p/224, k=seg*200+p%224
__global__ void k_cvt_w(const float* __restrict__ src, _Float16* __restrict__ dst,
                        int N, int Npad, int Korig, int Kpad) {
  int i = blockIdx.x * 256 + threadIdx.x;
  int total = Npad * Kpad;
  if (i >= total) return;
  int n = i / Kpad, p = i % Kpad;
  int seg = p / PF, kk = p % PF;
  int k = seg * F_ + kk;
  float v = 0.f;
  if (n < N && kk < F_ && k < Korig) v = src[(size_t)n * Korig + k];
  dst[i] = (_Float16)v;
}

__global__ void k_cvt_mol(const float* __restrict__ src, _Float16* __restrict__ dst) {
  int i = blockIdx.x * 256 + threadIdx.x;           // BDIM*PF
  if (i >= BDIM * PF) return;
  int b = i / PF, f = i % PF;
  dst[i] = (_Float16)((f < F_) ? src[b * F_ + f] : 0.f);
}

// --------------------------- phase 1 init ----------------------------------
// mw = softmax_L(<mol,act_feat>);  git = mw*mol + act_feat
__global__ __launch_bounds__(256) void k_phase1(
    const float* __restrict__ molF, const float* __restrict__ actv,
    float* __restrict__ stF, _Float16* __restrict__ stH) {
  __shared__ float red[256];
  int b = blockIdx.x, t = threadIdx.x;
  const float* mol = molF + (size_t)b * F_;
  const float* af  = actv + ((size_t)b * LDIM + t) * F_;
  float s = 0.f;
  for (int f = 0; f < F_; ++f) s += mol[f] * af[f];
  red[t] = s; __syncthreads();
  for (int o = 128; o > 0; o >>= 1) { if (t < o) red[t] = fmaxf(red[t], red[t + o]); __syncthreads(); }
  float mx = red[0]; __syncthreads();
  float e = expf(s - mx);
  red[t] = e; __syncthreads();
  for (int o = 128; o > 0; o >>= 1) { if (t < o) red[t] += red[t + o]; __syncthreads(); }
  float mw = e / red[0];
  size_t row = (size_t)b * LDIM + t;
  for (int f = 0; f < F_; ++f) {
    float g = mw * mol[f] + af[f];
    stF[row * F_ + f] = g;
    stH[row * PF + f] = (_Float16)g;
  }
}

// --------------------------- generic WMMA GEMM -----------------------------
// C[M,Npad] = act( A[M,Kpad] * W[Npad,Kpad]^T + bias ).
// One wave computes MBLK consecutive 16-row tiles at one 16-col tile, sharing
// each B (weight) fragment across MBLK WMMAs. Statically unrolled K loop.
// MODE 0: A row m = A0[m]                 (KSTEPS = 7)
// MODE 1: A row m = [ A0[m>>8] , A1[m] ]  (KSTEPS = 14)   -- mol concat
// MODE 2: A row m = [ A0[m>>3] , A0[(bl&~255)+idx[m]] ]   -- neighbor concat
template <int KSTEPS, int MODE, int DOELU, int MBLK>
__global__ __launch_bounds__(256) void k_gemm(
    const _Float16* __restrict__ A0, const _Float16* __restrict__ A1,
    const int* __restrict__ idx,
    const _Float16* __restrict__ W, const float* __restrict__ bias,
    float* __restrict__ outF, _Float16* __restrict__ outH,
    int M, int Norig, int Npad) {
  int wave = threadIdx.x >> 5;
  int lane = threadIdx.x & 31;
  int macroM = M / (16 * MBLK);
  int Ntiles = Npad >> 4;
  long tile = (long)blockIdx.x * 8 + wave;
  if (tile >= (long)macroM * Ntiles) return;
  int mt = (int)(tile % macroM);
  int nt = (int)(tile / macroM);
  int r = lane & 15, kh = lane >> 4;

  const _Float16 *p0[MBLK], *p1[MBLK];
#pragma unroll
  for (int i = 0; i < MBLK; ++i) {
    int m = (mt * MBLK + i) * 16 + r;
    if (MODE == 0)      { p0[i] = A0 + (size_t)m * PF;        p1[i] = p0[i]; }
    else if (MODE == 1) { p0[i] = A0 + (size_t)(m >> 8) * PF; p1[i] = A1 + (size_t)m * PF; }
    else {
      int bl = m >> 3; int nv = idx[m];
      p0[i] = A0 + (size_t)bl * PF;
      p1[i] = A0 + (size_t)((bl & ~(LDIM - 1)) + nv) * PF;
    }
  }
  const _Float16* wrow = W + (size_t)(nt * 16 + r) * (size_t)(KSTEPS * 32);

  v8f acc[MBLK];
#pragma unroll
  for (int i = 0; i < MBLK; ++i) { v8f z = {}; acc[i] = z; }

#pragma unroll
  for (int s = 0; s < KSTEPS; ++s) {
    v16h b = load_frag(wrow + s * 32 + kh * 8);
#pragma unroll
    for (int i = 0; i < MBLK; ++i) {
      const _Float16* ap = (s < 7) ? (p0[i] + s * 32) : (p1[i] + (s - 7) * 32);
      v16h a = load_frag(ap + kh * 8);
      acc[i] = wmma_f16(a, b, acc[i]);
    }
  }

  int c = nt * 16 + (lane & 15);
  if (c < Norig) {
    float bv = bias ? bias[c] : 0.f;
#pragma unroll
    for (int i = 0; i < MBLK; ++i) {
      int mbase = (mt * MBLK + i) * 16 + (lane >> 4) * 8;
#pragma unroll
      for (int v = 0; v < 8; ++v) {
        float val = acc[i][v] + bv;
        if (DOELU) val = eluf(val);
        int mr = mbase + v;
        if (outF) outF[(size_t)mr * Npad + c] = val;
        if (outH) outH[(size_t)mr * PF + c] = (_Float16)val;
      }
    }
  }
}

// --------------------------- fused attention round -------------------------
// one wave per (b,l): scores = leaky(concat(self,nbr)@Wal^T+b), pad-masked
// softmax over N=8 (lives in one lane's 8 acc VGPRs), nt = nbr@Wat^T+b,
// ctx = elu(sum_n attw*nt). Neighbor rows staged in LDS.
__global__ __launch_bounds__(128) void k_attn(
    const _Float16* __restrict__ stH, const int* __restrict__ deg,
    const _Float16* __restrict__ Wal, const float* __restrict__ alB,
    const _Float16* __restrict__ Wat, const float* __restrict__ atB,
    _Float16* __restrict__ ctxH) {
  __shared__ _Float16 sR[4][9][PF];       // [wave][nbr0..7, self=8][feat]
  int w = threadIdx.x >> 5, lane = threadIdx.x & 31;
  int bl = blockIdx.x * 4 + w;

  int idxv[8];
#pragma unroll
  for (int n = 0; n < 8; ++n) idxv[n] = deg[bl * 8 + n];
#pragma unroll
  for (int rowi = 0; rowi < 9; ++rowi) {
    int src = (rowi == 8) ? bl : ((bl & ~(LDIM - 1)) + idxv[rowi]);
    if (lane < 28)
      *(float4*)(&sR[w][rowi][lane * 8]) =
          *(const float4*)(stH + (size_t)src * PF + lane * 8);
  }
  __syncthreads();

  int r = lane & 15, kh = lane >> 4;
  int nb = r & 7;                          // rows 8..15 mirror 0..7 (unused)
  for (int ntile = 0; ntile < 13; ++ntile) {
    v8f accS = {};
    const _Float16* wrow = Wal + (size_t)(ntile * 16 + r) * 448;
#pragma unroll
    for (int s = 0; s < 14; ++s) {
      const _Float16* ap = (s < 7) ? (&sR[w][8][s * 32]) : (&sR[w][nb][(s - 7) * 32]);
      v16h a = load_frag(ap + kh * 8);
      v16h b = load_frag(wrow + s * 32 + kh * 8);
      accS = wmma_f16(a, b, accS);
    }
    v8f accT = {};
    const _Float16* wrow2 = Wat + (size_t)(ntile * 16 + r) * PF;
#pragma unroll
    for (int s = 0; s < 7; ++s) {
      v16h a = load_frag(&sR[w][nb][s * 32] + kh * 8);
      v16h b = load_frag(wrow2 + s * 32 + kh * 8);
      accT = wmma_f16(a, b, accT);
    }
    if (lane < 16) {                      // lanes 0..15 hold rows n=0..7
      int f = ntile * 16 + lane;
      float ab = (f < F_) ? alB[f] : 0.f;
      float tb = (f < F_) ? atB[f] : 0.f;
      float e[8], mx = -3.4e38f;
#pragma unroll
      for (int v = 0; v < 8; ++v) {
        float x = accS[v] + ab;
        x = x > 0.f ? x : 0.01f * x;       // leaky_relu
        if (idxv[v] == LDIM - 1) x = -9e8f; // softmax pad mask
        e[v] = x; mx = fmaxf(mx, x);
      }
      float sum = 0.f;
#pragma unroll
      for (int v = 0; v < 8; ++v) { e[v] = expf(e[v] - mx); sum += e[v]; }
      float inv = 1.f / sum;
      float ctx = 0.f;
#pragma unroll
      for (int v = 0; v < 8; ++v) {
        float av = (idxv[v] == LDIM - 1) ? 0.f : e[v] * inv; // attend mask
        ctx += av * (accT[v] + tb);
      }
      ctx = eluf(ctx);
      if (f < 208) ctxH[(size_t)bl * PF + f] = (_Float16)((f < F_) ? ctx : 0.f);
    }
  }
}

// --------------------------- GRU combine / relu ----------------------------
__global__ void k_gru(const float* __restrict__ gi, const float* __restrict__ gh,
                      float* __restrict__ stF, _Float16* __restrict__ stH, int doRelu) {
  int i = blockIdx.x * 256 + threadIdx.x;  // M_ATOMS*F_
  int m = i / F_, f = i % F_;
  size_t base = (size_t)m * 608;
  float rr = sigm(gi[base + f] + gh[base + f]);
  float zz = sigm(gi[base + F_ + f] + gh[base + F_ + f]);
  float nn = tanhf(gi[base + 2 * F_ + f] + rr * gh[base + 2 * F_ + f]);
  float h = (1.f - zz) * nn + zz * stF[(size_t)m * F_ + f];
  if (doRelu) h = fmaxf(h, 0.f);
  stF[(size_t)m * F_ + f] = h;
  stH[(size_t)m * PF + f] = (_Float16)h;
}

__global__ void k_relu(float* __restrict__ stF, _Float16* __restrict__ stH) {
  int i = blockIdx.x * 256 + threadIdx.x;  // M_ATOMS*F_
  int m = i / F_, f = i % F_;
  float h = fmaxf(stF[(size_t)m * F_ + f], 0.f);
  stF[(size_t)m * F_ + f] = h;
  stH[(size_t)m * PF + f] = (_Float16)h;
}

// --------------------------- decodes ---------------------------------------
__global__ void k_dec_atom(const float* __restrict__ al, float* __restrict__ out) {
  int m = blockIdx.x * 256 + threadIdx.x;  // M_ATOMS
  const float* a = al + (size_t)m * 48;
  float o[39];
#pragma unroll
  for (int i = 0; i < 39; ++i) o[i] = 0.f;
  const int lo[4] = {0, 16, 24, 31}, hi[4] = {16, 22, 30, 36};
  for (int sgi = 0; sgi < 4; ++sgi) {
    float mx = -3.4e38f;
    for (int i = lo[sgi]; i < hi[sgi]; ++i) mx = fmaxf(mx, a[i]);
    float sum = 0.f;
    for (int i = lo[sgi]; i < hi[sgi]; ++i) sum += expf(a[i] - mx);
    float inv = 1.f / sum;
    for (int i = lo[sgi]; i < hi[sgi]; ++i) o[i] += expf(a[i] - mx) * inv;
  }
  o[24] += fmaxf(a[24], 0.f);
  o[30] += sigm(a[30]);
  o[36] += sigm(a[36]);
  o[37] += sigm(a[37]);
  o[38] += sigm(a[38]);
  float* po = out + (size_t)m * 39;
#pragma unroll
  for (int i = 0; i < 39; ++i) po[i] = o[i];
}

__global__ void k_dec_bond(const float* __restrict__ bl, float* __restrict__ out) {
  int m = blockIdx.x * 256 + threadIdx.x;  // M_BONDS
  const float* a = bl + (size_t)m * 16;
  float o[10];
#pragma unroll
  for (int i = 0; i < 10; ++i) o[i] = 0.f;
  const int lo[2] = {0, 6}, hi[2] = {4, 10};
  for (int sgi = 0; sgi < 2; ++sgi) {
    float mx = -3.4e38f;
    for (int i = lo[sgi]; i < hi[sgi]; ++i) mx = fmaxf(mx, a[i]);
    float sum = 0.f;
    for (int i = lo[sgi]; i < hi[sgi]; ++i) sum += expf(a[i] - mx);
    float inv = 1.f / sum;
    for (int i = lo[sgi]; i < hi[sgi]; ++i) o[i] += expf(a[i] - mx) * inv;
  }
  o[4] += sigm(a[4]);
  o[5] += sigm(a[5]);
  float* po = out + (size_t)m * 10;
#pragma unroll
  for (int i = 0; i < 10; ++i) po[i] = o[i];
}

// ---------------------------------------------------------------------------
extern "C" void kernel_launch(void* const* d_in, const int* in_sizes, int n_in,
                              void* d_out, int out_size, void* d_ws, size_t ws_size,
                              hipStream_t stream) {
  (void)in_sizes; (void)n_in; (void)out_size; (void)ws_size;
  const int*   deg        = (const int*)  d_in[2];
  const float* mol        = (const float*)d_in[5];
  const float* actv       = (const float*)d_in[6];
  const float* atom_fc_w  = (const float*)d_in[7];
  const float* atom_fc_b  = (const float*)d_in[8];
  const float* bond_fc_w  = (const float*)d_in[9];
  const float* bond_fc_b  = (const float*)d_in[10];
  const float* align_w    = (const float*)d_in[11];
  const float* align_b    = (const float*)d_in[12];
  const float* attend_w   = (const float*)d_in[13];
  const float* attend_b   = (const float*)d_in[14];
  const float* gru_wih    = (const float*)d_in[15];
  const float* gru_whh    = (const float*)d_in[16];
  const float* gru_bih    = (const float*)d_in[17];
  const float* gru_bhh    = (const float*)d_in[18];
  const float* mol_al_w   = (const float*)d_in[19];
  const float* mol_al_b   = (const float*)d_in[20];
  const float* mol_gih    = (const float*)d_in[21];
  const float* mol_ghh    = (const float*)d_in[22];
  const float* mol_bih    = (const float*)d_in[23];
  const float* mol_bhh    = (const float*)d_in[24];

  // ---- workspace bump allocator (total ~218 MB) ----
  char* ws = (char*)d_ws;
  size_t off = 0;
  auto take = [&](size_t bytes) -> void* {
    void* p = ws + off;
    off = (off + bytes + 255) & ~(size_t)255;
    return p;
  };
  _Float16* wMolAl  = (_Float16*)take((size_t)208 * 448 * 2);
  _Float16* wMolGih = (_Float16*)take((size_t)608 * 224 * 2);
  _Float16* wMolGhh = (_Float16*)take((size_t)608 * 224 * 2);
  _Float16* wAlign[2], *wAttend[2], *wGih[2], *wGhh[2];
  for (int d = 0; d < 2; ++d) wAlign[d]  = (_Float16*)take((size_t)208 * 448 * 2);
  for (int d = 0; d < 2; ++d) wAttend[d] = (_Float16*)take((size_t)208 * 224 * 2);
  for (int d = 0; d < 2; ++d) wGih[d]    = (_Float16*)take((size_t)608 * 224 * 2);
  for (int d = 0; d < 2; ++d) wGhh[d]    = (_Float16*)take((size_t)608 * 224 * 2);
  _Float16* wAtomFc = (_Float16*)take((size_t)48 * 224 * 2);
  _Float16* wBondFc = (_Float16*)take((size_t)16 * 448 * 2);
  _Float16* molH = (_Float16*)take((size_t)BDIM * PF * 2);
  _Float16* stH  = (_Float16*)take((size_t)M_ATOMS * PF * 2);
  _Float16* ctxH = (_Float16*)take((size_t)M_ATOMS * PF * 2);
  float* stF = (float*)take((size_t)M_ATOMS * F_ * 4);
  float* giF = (float*)take((size_t)M_ATOMS * 608 * 4);
  float* ghF = (float*)take((size_t)M_ATOMS * 608 * 4);
  float* alF = giF;   // alias: decode-time reuse (32768*48 <= 32768*608)
  float* blF = ghF;   // alias: decode-time reuse (262144*16 <= 32768*608)

  hipMemsetAsync(stH,  0, (size_t)M_ATOMS * PF * 2, stream);
  hipMemsetAsync(ctxH, 0, (size_t)M_ATOMS * PF * 2, stream);

  auto cvt = [&](const float* s, _Float16* dst, int N, int Npad, int K, int Kp) {
    int total = Npad * Kp;
    k_cvt_w<<<(total + 255) / 256, 256, 0, stream>>>(s, dst, N, Npad, K, Kp);
  };
  cvt(mol_al_w, wMolAl, 200, 208, 400, 448);
  cvt(mol_gih,  wMolGih, 600, 608, 200, 224);
  cvt(mol_ghh,  wMolGhh, 600, 608, 200, 224);
  for (int d = 0; d < 2; ++d) cvt(align_w  + (size_t)d * 200 * 400, wAlign[d],  200, 208, 400, 448);
  for (int d = 0; d < 2; ++d) cvt(attend_w + (size_t)d * 200 * 200, wAttend[d], 200, 208, 200, 224);
  for (int d = 0; d < 2; ++d) cvt(gru_wih  + (size_t)d * 600 * 200, wGih[d],    600, 608, 200, 224);
  for (int d = 0; d < 2; ++d) cvt(gru_whh  + (size_t)d * 600 * 200, wGhh[d],    600, 608, 200, 224);
  cvt(atom_fc_w, wAtomFc, 39, 48, 200, 224);
  cvt(bond_fc_w, wBondFc, 10, 16, 400, 448);
  k_cvt_mol<<<(BDIM * PF + 255) / 256, 256, 0, stream>>>(mol, molH);

  // phase 1: mw softmax + git init
  k_phase1<<<BDIM, 256, 0, stream>>>(mol, actv, stF, stH);

  const int gElem = (M_ATOMS * F_) / 256;                     // 25600
  const int mAcro = M_ATOMS / 64;                             // 512 macro M-tiles
  const int gCtx  = (mAcro * 13 + 7) / 8;                     // 832
  const int gGru  = (mAcro * 38 + 7) / 8;                     // 2432
  const int gAFc  = (mAcro * 3 + 7) / 8;                      // 192
  const int gBFc  = ((M_BONDS / 64) * 1 + 7) / 8;             // 512

  // mol GRU, T = 2 steps
  for (int t = 0; t < 2; ++t) {
    k_gemm<14, 1, 1, 4><<<gCtx, 256, 0, stream>>>(molH, stH, nullptr, wMolAl, mol_al_b,
                                                  nullptr, ctxH, M_ATOMS, 200, 208);
    k_gemm<7, 0, 0, 4><<<gGru, 256, 0, stream>>>(ctxH, nullptr, nullptr, wMolGih, mol_bih,
                                                 giF, nullptr, M_ATOMS, 600, 608);
    k_gemm<7, 0, 0, 4><<<gGru, 256, 0, stream>>>(stH, nullptr, nullptr, wMolGhh, mol_bhh,
                                                 ghF, nullptr, M_ATOMS, 600, 608);
    k_gru<<<gElem, 256, 0, stream>>>(giF, ghF, stF, stH, 0);
  }
  k_relu<<<gElem, 256, 0, stream>>>(stF, stH);

  // attention rounds, R-1 = 2
  for (int d = 0; d < 2; ++d) {
    k_attn<<<M_ATOMS / 4, 128, 0, stream>>>(stH, deg, wAlign[d], align_b + d * 200,
                                            wAttend[d], attend_b + d * 200, ctxH);
    k_gemm<7, 0, 0, 4><<<gGru, 256, 0, stream>>>(ctxH, nullptr, nullptr, wGih[d], gru_bih + d * 600,
                                                 giF, nullptr, M_ATOMS, 600, 608);
    k_gemm<7, 0, 0, 4><<<gGru, 256, 0, stream>>>(stH, nullptr, nullptr, wGhh[d], gru_bhh + d * 600,
                                                 ghF, nullptr, M_ATOMS, 600, 608);
    k_gru<<<gElem, 256, 0, stream>>>(giF, ghF, stF, stH, 1);
  }

  // decode GEMMs
  k_gemm<7, 0, 0, 4><<<gAFc, 256, 0, stream>>>(stH, nullptr, nullptr, wAtomFc, atom_fc_b,
                                               alF, nullptr, M_ATOMS, 39, 48);
  k_gemm<14, 2, 0, 4><<<gBFc, 256, 0, stream>>>(stH, nullptr, deg, wBondFc, bond_fc_b,
                                                blF, nullptr, M_BONDS, 10, 16);

  // activations -> d_out (atoms first, then bonds)
  float* out = (float*)d_out;
  k_dec_atom<<<M_ATOMS / 256, 256, 0, stream>>>(alF, out);
  k_dec_bond<<<M_BONDS / 256, 256, 0, stream>>>(blF, out + (size_t)M_ATOMS * 39);
}
